// AttnDecoderRNN_16501264351732
// MI455X (gfx1250) — matrix operand order repaired and checked by reading, not compiled, not run on hardware
//
#include <hip/hip_runtime.h>
#include <hip/hip_bf16.h>

// Problem constants (match reference)
#define Bb  32
#define Ss  64
#define Tt  32
#define Hh  512
#define Ll  2
#define Vv  32000
#define BOSTOK 1

typedef __attribute__((ext_vector_type(16))) __bf16 bf16x16;
typedef __attribute__((ext_vector_type(8)))  float  f32x8;

struct U32x8 { uint4 lo, hi; };

__device__ __forceinline__ unsigned short f2bf(float f) {
  unsigned u = __float_as_uint(f);
  u += 0x7FFFu + ((u >> 16) & 1u);      // round-to-nearest-even
  return (unsigned short)(u >> 16);
}
__device__ __forceinline__ float sigmoidf(float x) { return 1.0f / (1.0f + expf(-x)); }
__device__ __forceinline__ float dot4(float4 a, float4 b) {
  return a.x * b.x + a.y * b.y + a.z * b.z + a.w * b.w;
}

// ---------------- setup kernels ----------------

__global__ void k_init(const float* __restrict__ ehid, float* h0, float* h1, int* tok) {
  int i = blockIdx.x * blockDim.x + threadIdx.x;
  if (i < Bb * Hh)          h0[i]            = ehid[i];
  else if (i < 2 * Bb * Hh) h1[i - Bb * Hh]  = ehid[i];
  if (i < Bb) tok[i] = BOSTOK;
}

__global__ void k_convw(const float* __restrict__ w, unsigned short* __restrict__ wb, int n) {
  int i = blockIdx.x * blockDim.x + threadIdx.x;
  if (i < n) wb[i] = f2bf(w[i]);
}

// Ukeys[b,s,i] = bua[i] + sum_j enc[b,s,j] * Ua[i,j]
__global__ void k_ukeys(const float* __restrict__ enc, const float* __restrict__ Ua,
                        const float* __restrict__ bua, float* __restrict__ Uk) {
  int id  = blockIdx.x * blockDim.x + threadIdx.x;   // row*512 + i
  int row = id >> 9, i = id & 511;
  const float4* e = (const float4*)(enc + (size_t)row * Hh);
  const float4* u = (const float4*)(Ua  + (size_t)i * Hh);
  float acc = bua[i];
  for (int j = 0; j < Hh / 4; ++j) acc += dot4(e[j], u[j]);
  Uk[id] = acc;
}

// ---------------- per-step kernels ----------------

// q[b,i] = ba[i] + [h0_b ; h1_b] . Wa[i,:]
__global__ void k_q(const float* __restrict__ h0, const float* __restrict__ h1,
                    const float* __restrict__ Wa, const float* __restrict__ ba,
                    float* __restrict__ q) {
  int id = blockIdx.x * blockDim.x + threadIdx.x;
  int b = id >> 9, i = id & 511;
  const float4* w  = (const float4*)(Wa + (size_t)i * (Ll * Hh));
  const float4* a0 = (const float4*)(h0 + (size_t)b * Hh);
  const float4* a1 = (const float4*)(h1 + (size_t)b * Hh);
  float acc = ba[i];
  for (int j = 0; j < Hh / 4; ++j) acc += dot4(a0[j], w[j]);
  for (int j = 0; j < Hh / 4; ++j) acc += dot4(a1[j], w[Hh / 4 + j]);
  q[id] = acc;
}

// scores[b,s] = bva + sum_i Va[i]*tanh(q[b,i]+Uk[b,s,i]) ; one block per (b,s)
__global__ void k_scores(const float* __restrict__ q, const float* __restrict__ Uk,
                         const float* __restrict__ Va, const float* __restrict__ bva,
                         float* __restrict__ sc) {
  int bs = blockIdx.x;
  int b  = bs >> 6;
  const float* qq = q  + (size_t)b  * Hh;
  const float* uk = Uk + (size_t)bs * Hh;
  float p = 0.f;
  for (int i = threadIdx.x; i < Hh; i += 256) p += Va[i] * tanhf(qq[i] + uk[i]);
  __shared__ float red[256];
  red[threadIdx.x] = p; __syncthreads();
  for (int s = 128; s > 0; s >>= 1) {
    if ((int)threadIdx.x < s) red[threadIdx.x] += red[threadIdx.x + s];
    __syncthreads();
  }
  if (threadIdx.x == 0) sc[bs] = red[0] + bva[0];
}

// softmax over S, write attn weights, context, and gin = [emb[tok] ; context]; one block per b
__global__ void k_attn(const float* __restrict__ sc, const float* __restrict__ enc,
                       const float* __restrict__ emb, const int* __restrict__ tok,
                       float* __restrict__ gin, float* __restrict__ outAttn, int t) {
  int b = blockIdx.x;
  __shared__ float w[Ss];
  __shared__ float tmp[Ss];
  if (threadIdx.x < Ss) tmp[threadIdx.x] = sc[b * Ss + threadIdx.x];
  __syncthreads();
  float mx = -3.4e38f;
  for (int s = 0; s < Ss; ++s) mx = fmaxf(mx, tmp[s]);
  float sum = 0.f;
  for (int s = 0; s < Ss; ++s) sum += expf(tmp[s] - mx);
  float inv = 1.0f / sum;
  if (threadIdx.x < Ss) {
    float ws = expf(tmp[threadIdx.x] - mx) * inv;
    w[threadIdx.x] = ws;
    outAttn[((size_t)b * Tt + t) * Ss + threadIdx.x] = ws;
  }
  __syncthreads();
  const float* eb = enc + (size_t)b * Ss * Hh;
  int tk = tok[b];
  for (int h = threadIdx.x; h < Hh; h += 256) {
    float c = 0.f;
    for (int s = 0; s < Ss; ++s) c += w[s] * eb[(size_t)s * Hh + h];
    gin[(size_t)b * (2 * Hh) + Hh + h] = c;
    gin[(size_t)b * (2 * Hh) + h]      = emb[(size_t)tk * Hh + h];
  }
}

// GRU0: input gin [B,2H], hidden h0 -> h0n  (torch gate order r,z,n)
__global__ void k_gru0(const float* __restrict__ gin, const float* __restrict__ h0,
                       const float* __restrict__ Wih, const float* __restrict__ Whh,
                       const float* __restrict__ bih, const float* __restrict__ bhh,
                       float* __restrict__ h0n) {
  int id = blockIdx.x * blockDim.x + threadIdx.x;
  int b = id >> 9, j = id & 511;
  const float4* x = (const float4*)(gin + (size_t)b * (2 * Hh));
  const float4* h = (const float4*)(h0  + (size_t)b * Hh);
  float ir = bih[j], iz = bih[Hh + j], in_ = bih[2 * Hh + j];
  const float4* wr = (const float4*)(Wih + (size_t)j            * (2 * Hh));
  const float4* wz = (const float4*)(Wih + (size_t)(Hh + j)     * (2 * Hh));
  const float4* wn = (const float4*)(Wih + (size_t)(2 * Hh + j) * (2 * Hh));
  for (int k = 0; k < (2 * Hh) / 4; ++k) {
    float4 xv = x[k];
    ir += dot4(xv, wr[k]); iz += dot4(xv, wz[k]); in_ += dot4(xv, wn[k]);
  }
  float hr = bhh[j], hz = bhh[Hh + j], hn = bhh[2 * Hh + j];
  const float4* vr = (const float4*)(Whh + (size_t)j            * Hh);
  const float4* vz = (const float4*)(Whh + (size_t)(Hh + j)     * Hh);
  const float4* vn = (const float4*)(Whh + (size_t)(2 * Hh + j) * Hh);
  for (int k = 0; k < Hh / 4; ++k) {
    float4 hv = h[k];
    hr += dot4(hv, vr[k]); hz += dot4(hv, vz[k]); hn += dot4(hv, vn[k]);
  }
  float r = sigmoidf(ir + hr), z = sigmoidf(iz + hz);
  float n = tanhf(in_ + r * hn);
  h0n[id] = (1.f - z) * n + z * h0[(size_t)b * Hh + j];
}

// GRU1: input h0n [B,H], hidden h1 -> h1n ; also commits h0 = h0n
__global__ void k_gru1(const float* __restrict__ h0n, const float* __restrict__ h1,
                       const float* __restrict__ Wih, const float* __restrict__ Whh,
                       const float* __restrict__ bih, const float* __restrict__ bhh,
                       float* __restrict__ h1n, float* __restrict__ h0) {
  int id = blockIdx.x * blockDim.x + threadIdx.x;
  int b = id >> 9, j = id & 511;
  const float4* x = (const float4*)(h0n + (size_t)b * Hh);
  const float4* h = (const float4*)(h1  + (size_t)b * Hh);
  float ir = bih[j], iz = bih[Hh + j], in_ = bih[2 * Hh + j];
  const float4* wr = (const float4*)(Wih + (size_t)j            * Hh);
  const float4* wz = (const float4*)(Wih + (size_t)(Hh + j)     * Hh);
  const float4* wn = (const float4*)(Wih + (size_t)(2 * Hh + j) * Hh);
  for (int k = 0; k < Hh / 4; ++k) {
    float4 xv = x[k];
    ir += dot4(xv, wr[k]); iz += dot4(xv, wz[k]); in_ += dot4(xv, wn[k]);
  }
  float hr = bhh[j], hz = bhh[Hh + j], hn = bhh[2 * Hh + j];
  const float4* vr = (const float4*)(Whh + (size_t)j            * Hh);
  const float4* vz = (const float4*)(Whh + (size_t)(Hh + j)     * Hh);
  const float4* vn = (const float4*)(Whh + (size_t)(2 * Hh + j) * Hh);
  for (int k = 0; k < Hh / 4; ++k) {
    float4 hv = h[k];
    hr += dot4(hv, vr[k]); hz += dot4(hv, vz[k]); hn += dot4(hv, vn[k]);
  }
  float r = sigmoidf(ir + hr), z = sigmoidf(iz + hz);
  float n = tanhf(in_ + r * hn);
  h1n[id] = (1.f - z) * n + z * h1[(size_t)b * Hh + j];
  h0[id]  = h0n[id];
}

__global__ void k_commit(const float* __restrict__ h1n, float* __restrict__ h1,
                         unsigned short* __restrict__ h1bf) {
  int i = blockIdx.x * blockDim.x + threadIdx.x;
  float v = h1n[i];
  h1[i] = v;
  h1bf[i] = f2bf(v);
}

// logits[b,v] = bf16-WMMA(h1, outW^T) + outb[v], written straight into d_out log-prob slot.
// One wave per 16-wide N strip, BOTH 16-row M-tiles per wave -> each outW fragment is
// loaded once and feeds two WMMAs (halves the dominant L2 weight traffic).
// 2000 waves = 250 blocks x 8 waves, every wave full (EXEC all ones).
__global__ void k_logits(const unsigned short* __restrict__ h1bf,
                         const unsigned short* __restrict__ wbf,
                         const float* __restrict__ outb,
                         float* __restrict__ outLP, int t) {
  int wave  = (blockIdx.x * blockDim.x + threadIdx.x) >> 5;   // = ntile
  int lane  = threadIdx.x & 31;
  int half  = lane >> 4;     // K-half selector
  int lm    = lane & 15;
  int n = wave * 16 + lm;    // B column for this lane
  // A rows for this lane: M-tile0 -> row lm, M-tile1 -> row 16+lm
  const unsigned short* arow0 = h1bf + (size_t)lm * Hh;
  const unsigned short* arow1 = h1bf + (size_t)(16 + lm) * Hh;
  const unsigned short* brow  = wbf  + (size_t)n * Hh + half * 8;
  f32x8 acc0 = {};
  f32x8 acc1 = {};
#pragma unroll 2
  for (int k0 = 0; k0 < Hh; k0 += 32) {
    __builtin_prefetch(brow + k0 + 64, 0, 1);   // global_prefetch_b8 on the B stream
    U32x8 ua0, ua1, ub;
    // A 16-bit 16x32 layout: VGPR0-3 = K {0..7 | 8..15}+half*8 ; VGPR4-7 = +16
    ua0.lo = *(const uint4*)(arow0 + k0 + half * 8);
    ua0.hi = *(const uint4*)(arow0 + k0 + 16 + half * 8);
    ua1.lo = *(const uint4*)(arow1 + k0 + half * 8);
    ua1.hi = *(const uint4*)(arow1 + k0 + 16 + half * 8);
    // B 16-bit 32x16 layout: same K order as A: K {0..7 | 8..15}+half*8 ; then +16
    ub.lo = *(const uint4*)(brow + k0);
    ub.hi = *(const uint4*)(brow + k0 + 16);
    bf16x16 a0 = __builtin_bit_cast(bf16x16, ua0);
    bf16x16 a1 = __builtin_bit_cast(bf16x16, ua1);
    bf16x16 bm = __builtin_bit_cast(bf16x16, ub);
    acc0 = __builtin_amdgcn_wmma_f32_16x16x32_bf16(false, a0, false, bm, (short)0, acc0,
                                                   false, false);
    acc1 = __builtin_amdgcn_wmma_f32_16x16x32_bf16(false, a1, false, bm, (short)0, acc1,
                                                   false, false);
  }
  float bias = outb[n];
  // C/D f32 16x16 layout: VGPR r -> M = r + half*8, N = lm
#pragma unroll
  for (int r = 0; r < 8; ++r) {
    int mm = half * 8 + r;
    outLP[(size_t)mm * Tt * Vv + (size_t)t * Vv + n]        = acc0[r] + bias;
    outLP[(size_t)(16 + mm) * Tt * Vv + (size_t)t * Vv + n] = acc1[r] + bias;
  }
}

// first-occurrence argmax over V per batch row -> next token
__global__ void k_argmax(const float* __restrict__ outLP, int* __restrict__ tok, int t) {
  int b = blockIdx.x;
  const float* row = outLP + (size_t)b * Tt * Vv + (size_t)t * Vv;
  float bv = -3.4e38f; int bi = 0;
  for (int v = threadIdx.x; v < Vv; v += 256) {
    float x = row[v];
    if (x > bv) { bv = x; bi = v; }
  }
  __shared__ float sv[256];
  __shared__ int   si[256];
  sv[threadIdx.x] = bv; si[threadIdx.x] = bi; __syncthreads();
  for (int s = 128; s > 0; s >>= 1) {
    if ((int)threadIdx.x < s) {
      float ov = sv[threadIdx.x + s]; int oi = si[threadIdx.x + s];
      if (ov > sv[threadIdx.x] || (ov == sv[threadIdx.x] && oi < si[threadIdx.x])) {
        sv[threadIdx.x] = ov; si[threadIdx.x] = oi;
      }
    }
    __syncthreads();
  }
  if (threadIdx.x == 0) tok[b] = si[0];
}

__global__ void k_hidout(const float* __restrict__ h0, const float* __restrict__ h1,
                         float* __restrict__ out) {
  int i = blockIdx.x * blockDim.x + threadIdx.x;
  if (i < Bb * Hh)          out[i] = h0[i];
  else if (i < 2 * Bb * Hh) out[i] = h1[i - Bb * Hh];
}

// in-place log-softmax over each (b,t) row of V
__global__ void k_lsm(float* __restrict__ lp) {
  float* x = lp + (size_t)blockIdx.x * Vv;
  __shared__ float red[256];
  float mx = -3.4e38f;
  for (int v = threadIdx.x; v < Vv; v += 256) mx = fmaxf(mx, x[v]);
  red[threadIdx.x] = mx; __syncthreads();
  for (int s = 128; s > 0; s >>= 1) {
    if ((int)threadIdx.x < s) red[threadIdx.x] = fmaxf(red[threadIdx.x], red[threadIdx.x + s]);
    __syncthreads();
  }
  mx = red[0]; __syncthreads();
  float sum = 0.f;
  for (int v = threadIdx.x; v < Vv; v += 256) sum += expf(x[v] - mx);
  red[threadIdx.x] = sum; __syncthreads();
  for (int s = 128; s > 0; s >>= 1) {
    if ((int)threadIdx.x < s) red[threadIdx.x] += red[threadIdx.x + s];
    __syncthreads();
  }
  float lse = mx + logf(red[0]);
  for (int v = threadIdx.x; v < Vv; v += 256) x[v] = x[v] - lse;
}

// ---------------- host-side launch ----------------

extern "C" void kernel_launch(void* const* d_in, const int* in_sizes, int n_in,
                              void* d_out, int out_size, void* d_ws, size_t ws_size,
                              hipStream_t stream) {
  const float* enc  = (const float*)d_in[0];
  const float* ehid = (const float*)d_in[1];
  /* d_in[2] target_tensor unused by the math */
  const float* emb  = (const float*)d_in[3];
  const float* Wa   = (const float*)d_in[4];
  const float* ba   = (const float*)d_in[5];
  const float* Ua   = (const float*)d_in[6];
  const float* bua  = (const float*)d_in[7];
  const float* Va   = (const float*)d_in[8];
  const float* bva  = (const float*)d_in[9];
  const float* W0ih = (const float*)d_in[10];
  const float* W0hh = (const float*)d_in[11];
  const float* b0ih = (const float*)d_in[12];
  const float* b0hh = (const float*)d_in[13];
  const float* W1ih = (const float*)d_in[14];
  const float* W1hh = (const float*)d_in[15];
  const float* b1ih = (const float*)d_in[16];
  const float* b1hh = (const float*)d_in[17];
  const float* outW = (const float*)d_in[18];
  const float* outb = (const float*)d_in[19];
  float* out = (float*)d_out;

  // workspace bump allocator (256B aligned)
  char* p = (char*)d_ws;
  auto alloc = [&](size_t bytes) -> char* {
    char* r = p;
    p += (bytes + 255) & ~(size_t)255;
    return r;
  };
  unsigned short* wbf  = (unsigned short*)alloc(sizeof(unsigned short) * (size_t)Vv * Hh);
  float*          Uk   = (float*)alloc(sizeof(float) * (size_t)Bb * Ss * Hh);
  float*          h0   = (float*)alloc(sizeof(float) * Bb * Hh);
  float*          h1   = (float*)alloc(sizeof(float) * Bb * Hh);
  float*          h0n  = (float*)alloc(sizeof(float) * Bb * Hh);
  float*          h1n  = (float*)alloc(sizeof(float) * Bb * Hh);
  unsigned short* h1bf = (unsigned short*)alloc(sizeof(unsigned short) * Bb * Hh);
  float*          qbuf = (float*)alloc(sizeof(float) * Bb * Hh);
  float*          sc   = (float*)alloc(sizeof(float) * Bb * Ss);
  float*          gin  = (float*)alloc(sizeof(float) * Bb * 2 * Hh);
  int*            tok  = (int*)alloc(sizeof(int) * Bb);

  float* lp   = out;                               // [B,T,V]
  float* hidO = out + (size_t)Bb * Tt * Vv;        // [L,B,H]
  float* attO = hidO + (size_t)Ll * Bb * Hh;       // [B,T,S]

  k_init <<<(2 * Bb * Hh + 255) / 256, 256, 0, stream>>>(ehid, h0, h1, tok);
  k_convw<<<((size_t)Vv * Hh) / 256, 256, 0, stream>>>(outW, wbf, Vv * Hh);
  k_ukeys<<<(Bb * Ss * Hh) / 256, 256, 0, stream>>>(enc, Ua, bua, Uk);

  for (int t = 0; t < Tt; ++t) {
    k_q     <<<(Bb * Hh) / 256, 256, 0, stream>>>(h0, h1, Wa, ba, qbuf);
    k_scores<<<Bb * Ss, 256, 0, stream>>>(qbuf, Uk, Va, bva, sc);
    k_attn  <<<Bb, 256, 0, stream>>>(sc, enc, emb, tok, gin, attO, t);
    k_gru0  <<<(Bb * Hh) / 256, 256, 0, stream>>>(gin, h0, W0ih, W0hh, b0ih, b0hh, h0n);
    k_gru1  <<<(Bb * Hh) / 256, 256, 0, stream>>>(h0n, h1, W1ih, W1hh, b1ih, b1hh, h1n, h0);
    k_commit<<<(Bb * Hh) / 256, 256, 0, stream>>>(h1n, h1, h1bf);
    // 2000 waves total, 8 waves/block -> 250 blocks, both M-tiles per wave
    k_logits<<<(Vv / 16) / 8, 256, 0, stream>>>(h1bf, wbf, outb, lp, t);
    k_argmax<<<Bb, 256, 0, stream>>>(lp, tok, t);
  }

  k_hidout<<<(2 * Bb * Hh + 255) / 256, 256, 0, stream>>>(h0, h1, hidO);
  k_lsm   <<<Bb * Tt, 256, 0, stream>>>(lp);
}
